// FinalModel_5557687681707
// MI455X (gfx1250) — matrix operand order repaired and checked by reading, not compiled, not run on hardware
//
#include <hip/hip_runtime.h>
#include <hip/hip_bf16.h>

// LocalMaskedMHCA for MI455X (gfx1250, wave32, WMMA + async-LDS).
// B=8, C=512, T=4096, H=8, HS=64, WIN=19, WOV=9
#define B_   8
#define C_   512
#define T_   4096
#define H_   8
#define HS_  64
#define WIN_ 19
#define WOV_ 9
#define SCALE_ 0.125f  // 1/sqrt(64)

typedef __bf16 v8bf  __attribute__((ext_vector_type(8)));
typedef __bf16 v16bf __attribute__((ext_vector_type(16)));
typedef float  v8f   __attribute__((ext_vector_type(8)));

#if __has_builtin(__builtin_amdgcn_global_load_async_to_lds_b32) && \
    __has_builtin(__builtin_amdgcn_s_wait_asynccnt)
#define USE_ASYNC_LDS 1
typedef __attribute__((address_space(1))) int gint_t;  // global int*
typedef __attribute__((address_space(3))) int lint_t;  // LDS int*
#endif

__device__ __forceinline__ unsigned short f2bf_u(float f) {
  unsigned u = __builtin_bit_cast(unsigned, f);
  u += 0x7fffu + ((u >> 16) & 1u);     // round-to-nearest-even
  return (unsigned short)(u >> 16);
}

// ---------------------------------------------------------------------------
// Kernel A: fused dwconv3 + BN-fold + 1x1 projection (Q/K/V) via WMMA.
// Output layout (B,T,C). Block: 256 thr = 8 waves; block tile 64(o) x 128(t).
// Wave w: o-subtile (w&3), t-range (w>>2)*64; A-fragment reused over 4 WMMAs.
// ---------------------------------------------------------------------------
__global__ __launch_bounds__(256)
void qkv_wmma_kernel(const float* __restrict__ x,
                     const float* __restrict__ dw_q, const float* __restrict__ dw_k,
                     const float* __restrict__ dw_v,
                     const float* __restrict__ bnq_s, const float* __restrict__ bnq_b,
                     const float* __restrict__ bnk_s, const float* __restrict__ bnk_b,
                     const float* __restrict__ bnv_s, const float* __restrict__ bnv_b,
                     const float* __restrict__ Wq, const float* __restrict__ bq,
                     const float* __restrict__ Wk, const float* __restrict__ bk,
                     const float* __restrict__ Wv, const float* __restrict__ bv,
                     float* __restrict__ qws, float* __restrict__ kws,
                     float* __restrict__ vws) {
  __shared__ __align__(16) unsigned short As[64 * 40];   // W tile  (o x c), bf16
  __shared__ __align__(16) unsigned short Bs[128 * 40];  // H tile^T (t x c), bf16

  const int tid   = threadIdx.x;
  const int lane  = tid & 31;
  const int wave  = tid >> 5;
  const int osub  = wave & 3;    // 0..3 -> 16 o each
  const int thalf = wave >> 2;   // 0..1 -> 64 t each

  const int t0 = blockIdx.x * 128;
  const int o0 = blockIdx.y * 64;
  const int b  = blockIdx.z / 3;
  const int br = blockIdx.z % 3;

  const float *dw, *bns, *bnb, *W, *bias;
  float* out;
  float oscale;
  if (br == 0)      { dw = dw_q; bns = bnq_s; bnb = bnq_b; W = Wq; bias = bq; out = qws; oscale = SCALE_; }
  else if (br == 1) { dw = dw_k; bns = bnk_s; bnb = bnk_b; W = Wk; bias = bk; out = kws; oscale = 1.0f; }
  else              { dw = dw_v; bns = bnv_s; bnb = bnv_b; W = Wv; bias = bv; out = vws; oscale = 1.0f; }

  v8f acc[4];
#pragma unroll
  for (int i = 0; i < 4; ++i)
    acc[i] = (v8f){0.f, 0.f, 0.f, 0.f, 0.f, 0.f, 0.f, 0.f};

  // CDNA5 fragment addressing (wave32):
  // A 16x32 bf16: lane L -> row M=L&15, K chunks [kb,kb+8) and [kb+16,kb+24), kb=(L>>4)*8
  // B 32x16 bf16: lane L -> col N=L&15, K chunks [kb,kb+8) and [kb+16,kb+24), kb=(L>>4)*8
  const int arow = osub * 16 + (lane & 15);
  const int akb  = (lane >> 4) * 8;
  const int browl = lane & 15;
  const int bkb   = (lane >> 4) * 8;

  for (int c0 = 0; c0 < C_; c0 += 32) {
    if (c0 + 32 < C_)  // uniform; pull next W tile toward the caches
      __builtin_prefetch(&W[(size_t)o0 * C_ + c0 + 32], 0, 0);

    // stage W[o0:+64, c0:+32] -> As (8 elems/thread)
#pragma unroll
    for (int k = 0; k < 8; ++k) {
      int idx = tid + k * 256;
      int r = idx >> 5, cc = idx & 31;
      As[r * 40 + cc] = f2bf_u(W[(size_t)(o0 + r) * C_ + c0 + cc]);
    }
    // stage H (dwconv3 + BN) transposed -> Bs[t_local][c_local] (16/thread)
#pragma unroll
    for (int k = 0; k < 16; ++k) {
      int idx = tid + k * 256;
      int tl = idx & 127, cl = idx >> 7;
      int c = c0 + cl, t = t0 + tl;
      const float* xr = x + ((size_t)b * C_ + c) * T_;
      float xm = (t > 0)      ? xr[t - 1] : 0.f;
      float xz = xr[t];
      float xp = (t < T_ - 1) ? xr[t + 1] : 0.f;
      float h = (xm * dw[c * 3 + 0] + xz * dw[c * 3 + 1] + xp * dw[c * 3 + 2]) * bns[c] + bnb[c];
      Bs[tl * 40 + cl] = f2bf_u(h);
    }
    __syncthreads();

    v8bf alo = *(const v8bf*)&As[arow * 40 + akb];
    v8bf ahi = *(const v8bf*)&As[arow * 40 + akb + 16];
    v16bf afrag = __builtin_shufflevector(alo, ahi, 0, 1, 2, 3, 4, 5, 6, 7,
                                          8, 9, 10, 11, 12, 13, 14, 15);
#pragma unroll
    for (int i = 0; i < 4; ++i) {
      int brow = thalf * 64 + i * 16 + browl;
      v8bf blo = *(const v8bf*)&Bs[brow * 40 + bkb];
      v8bf bhi = *(const v8bf*)&Bs[brow * 40 + bkb + 16];
      v16bf bfrag = __builtin_shufflevector(blo, bhi, 0, 1, 2, 3, 4, 5, 6, 7,
                                            8, 9, 10, 11, 12, 13, 14, 15);
      acc[i] = __builtin_amdgcn_wmma_f32_16x16x32_bf16(
          false, afrag, false, bfrag, (short)0, acc[i], false, false);
    }
    __syncthreads();
  }

  // D layout: VGPR r, lanes 0-15: M=r, lanes 16-31: M=r+8; N = lane&15
  const int ob = o0 + osub * 16 + ((lane >> 4) << 3);
  float bias8[8];
#pragma unroll
  for (int r = 0; r < 8; ++r) bias8[r] = bias[ob + r];
#pragma unroll
  for (int i = 0; i < 4; ++i) {
    const int t = t0 + thalf * 64 + i * 16 + (lane & 15);
    float* orow = out + ((size_t)b * T_ + t) * C_ + ob;
#pragma unroll
    for (int r = 0; r < 8; ++r)
      orow[r] = (acc[i][r] + bias8[r]) * oscale;
  }
}

// ---------------------------------------------------------------------------
// Kernel B: banded attention (window 19). One query per thread; K/V staged to
// LDS as fp32 via GLOBAL_LOAD_ASYNC_TO_LDS when available (ASYNCcnt path).
// ---------------------------------------------------------------------------
#define QB_  64
#define KR_  (QB_ + WIN_ - 1)  // 82 rows
#define KLDF 65                // float stride: 260B/row -> conflict-free

__global__ __launch_bounds__(QB_)
void local_attn_kernel(const float* __restrict__ qws, const float* __restrict__ kws,
                       const float* __restrict__ vws,
                       const unsigned char* __restrict__ mask,
                       const float* __restrict__ rel_pe,
                       float* __restrict__ attws) {
  __shared__ float kbuf[KR_ * KLDF];
  __shared__ float vbuf[KR_ * KLDF];
  __shared__ float sbuf[QB_ * WIN_];

  const int tid = threadIdx.x;
  const int t0  = blockIdx.x * QB_;
  const int h   = blockIdx.y;
  const int b   = blockIdx.z;

#ifdef USE_ASYNC_LDS
  // Async DMA of the K/V band into LDS (clamped addresses; edges zeroed after).
  for (int idx = tid; idx < KR_ * HS_; idx += QB_) {
    int row = idx >> 6, hs = idx & 63;
    int tr = t0 - WOV_ + row;
    int trc = tr < 0 ? 0 : (tr >= T_ ? T_ - 1 : tr);
    size_t base = ((size_t)b * T_ + trc) * C_ + h * HS_ + hs;
    __builtin_amdgcn_global_load_async_to_lds_b32(
        (gint_t*)(kws + base), (lint_t*)&kbuf[row * KLDF + hs], 0, 0);
    __builtin_amdgcn_global_load_async_to_lds_b32(
        (gint_t*)(vws + base), (lint_t*)&vbuf[row * KLDF + hs], 0, 0);
  }
  __builtin_amdgcn_s_wait_asynccnt(0);
  __syncthreads();
  if (t0 < WOV_ || t0 + QB_ + WOV_ > T_) {  // uniform: only edge blocks
    for (int idx = tid; idx < KR_ * HS_; idx += QB_) {
      int row = idx >> 6, hs = idx & 63;
      int tr = t0 - WOV_ + row;
      if (tr < 0 || tr >= T_) {
        kbuf[row * KLDF + hs] = 0.f;
        vbuf[row * KLDF + hs] = 0.f;
      }
    }
    __syncthreads();
  }
#else
  for (int idx = tid; idx < KR_ * HS_; idx += QB_) {
    int row = idx >> 6, hs = idx & 63;
    int tr = t0 - WOV_ + row;
    float kv = 0.f, vv = 0.f;
    if (tr >= 0 && tr < T_) {
      size_t base = ((size_t)b * T_ + tr) * C_ + h * HS_ + hs;
      kv = kws[base];
      vv = vws[base];
    }
    kbuf[row * KLDF + hs] = kv;
    vbuf[row * KLDF + hs] = vv;
  }
  __syncthreads();
#endif

  const int tq = t0 + tid;
  float qreg[HS_];
  {
    const float* qr = qws + ((size_t)b * T_ + tq) * C_ + h * HS_;
#pragma unroll
    for (int i = 0; i < HS_; ++i) qreg[i] = qr[i];
  }

  float* srow = &sbuf[tid * WIN_];
  const float NEG_INF = -__builtin_inff();
  for (int j = 0; j < WIN_; ++j) {
    const float* kr = &kbuf[(tid + j) * KLDF];
    float s = 0.f;
#pragma unroll
    for (int i = 0; i < HS_; ++i) s += qreg[i] * kr[i];
    s += rel_pe[h * WIN_ + j];
    int kidx = tq + j - WOV_;
    if (kidx < 0 || kidx >= T_) s = NEG_INF;
    else if (!mask[(size_t)b * T_ + kidx]) s += -1e4f;
    srow[j] = s;
  }

  float m = NEG_INF;
  for (int j = 0; j < WIN_; ++j) m = fmaxf(m, srow[j]);
  float sum = 0.f;
  for (int j = 0; j < WIN_; ++j) {
    float sj = srow[j];
    float e = (sj == NEG_INF) ? 0.f : __expf(sj - m);
    srow[j] = e;
    sum += e;
  }
  float qm  = mask[(size_t)b * T_ + tq] ? 1.f : 0.f;
  float inv = (sum > 0.f) ? (qm / sum) : 0.f;

  float accv[HS_];
#pragma unroll
  for (int i = 0; i < HS_; ++i) accv[i] = 0.f;
  for (int j = 0; j < WIN_; ++j) {
    float w = srow[j] * inv;
    const float* vr = &vbuf[(tid + j) * KLDF];
#pragma unroll
    for (int i = 0; i < HS_; ++i) accv[i] += w * vr[i];
  }

  float* orow = attws + ((size_t)b * T_ + tq) * C_ + h * HS_;
#pragma unroll
  for (int i = 0; i < HS_; ++i) orow[i] = accv[i];
}

// ---------------------------------------------------------------------------
// Kernel C: output 1x1 projection via WMMA. attws is (B,T,C): B-operand tile
// already (t,c) row-major. Block tile 64(o) x 128(t), 4 WMMAs per wave/K-step.
// Output (B,C,T) * mask.
// ---------------------------------------------------------------------------
__global__ __launch_bounds__(256)
void proj_wmma_kernel(const float* __restrict__ attws,
                      const float* __restrict__ Wp, const float* __restrict__ bp,
                      const unsigned char* __restrict__ mask,
                      float* __restrict__ out) {
  __shared__ __align__(16) unsigned short As[64 * 40];
  __shared__ __align__(16) unsigned short Bs[128 * 40];

  const int tid   = threadIdx.x;
  const int lane  = tid & 31;
  const int wave  = tid >> 5;
  const int osub  = wave & 3;
  const int thalf = wave >> 2;

  const int t0 = blockIdx.x * 128;
  const int o0 = blockIdx.y * 64;
  const int b  = blockIdx.z;

  v8f acc[4];
#pragma unroll
  for (int i = 0; i < 4; ++i)
    acc[i] = (v8f){0.f, 0.f, 0.f, 0.f, 0.f, 0.f, 0.f, 0.f};

  const int arow  = osub * 16 + (lane & 15);
  const int akb   = (lane >> 4) * 8;
  const int browl = lane & 15;
  const int bkb   = (lane >> 4) * 8;

  for (int c0 = 0; c0 < C_; c0 += 32) {
    if (c0 + 32 < C_)
      __builtin_prefetch(&Wp[(size_t)o0 * C_ + c0 + 32], 0, 0);
#pragma unroll
    for (int k = 0; k < 8; ++k) {
      int idx = tid + k * 256;
      int r = idx >> 5, cc = idx & 31;
      As[r * 40 + cc] = f2bf_u(Wp[(size_t)(o0 + r) * C_ + c0 + cc]);
    }
#pragma unroll
    for (int k = 0; k < 16; ++k) {
      int idx = tid + k * 256;
      int tl = idx >> 5, cl = idx & 31;
      Bs[tl * 40 + cl] =
          f2bf_u(attws[((size_t)b * T_ + t0 + tl) * C_ + c0 + cl]);
    }
    __syncthreads();

    v8bf alo = *(const v8bf*)&As[arow * 40 + akb];
    v8bf ahi = *(const v8bf*)&As[arow * 40 + akb + 16];
    v16bf afrag = __builtin_shufflevector(alo, ahi, 0, 1, 2, 3, 4, 5, 6, 7,
                                          8, 9, 10, 11, 12, 13, 14, 15);
#pragma unroll
    for (int i = 0; i < 4; ++i) {
      int brow = thalf * 64 + i * 16 + browl;
      v8bf blo = *(const v8bf*)&Bs[brow * 40 + bkb];
      v8bf bhi = *(const v8bf*)&Bs[brow * 40 + bkb + 16];
      v16bf bfrag = __builtin_shufflevector(blo, bhi, 0, 1, 2, 3, 4, 5, 6, 7,
                                            8, 9, 10, 11, 12, 13, 14, 15);
      acc[i] = __builtin_amdgcn_wmma_f32_16x16x32_bf16(
          false, afrag, false, bfrag, (short)0, acc[i], false, false);
    }
    __syncthreads();
  }

  const int ob = o0 + osub * 16 + ((lane >> 4) << 3);
  float bias8[8];
#pragma unroll
  for (int r = 0; r < 8; ++r) bias8[r] = bp[ob + r];
#pragma unroll
  for (int i = 0; i < 4; ++i) {
    const int t  = t0 + thalf * 64 + i * 16 + (lane & 15);
    const float mf = mask[(size_t)b * T_ + t] ? 1.f : 0.f;
#pragma unroll
    for (int r = 0; r < 8; ++r)
      out[((size_t)b * C_ + (ob + r)) * T_ + t] = (acc[i][r] + bias8[r]) * mf;
  }
}

// ---------------------------------------------------------------------------
extern "C" void kernel_launch(void* const* d_in, const int* in_sizes, int n_in,
                              void* d_out, int out_size, void* d_ws, size_t ws_size,
                              hipStream_t stream) {
  (void)in_sizes; (void)n_in; (void)out_size; (void)ws_size;
  const float* x            = (const float*)d_in[0];
  const unsigned char* mask = (const unsigned char*)d_in[1];
  const float* dw_q  = (const float*)d_in[2];
  const float* dw_k  = (const float*)d_in[3];
  const float* dw_v  = (const float*)d_in[4];
  const float* bnq_s = (const float*)d_in[5];
  const float* bnq_b = (const float*)d_in[6];
  const float* bnk_s = (const float*)d_in[7];
  const float* bnk_b = (const float*)d_in[8];
  const float* bnv_s = (const float*)d_in[9];
  const float* bnv_b = (const float*)d_in[10];
  const float* Wq = (const float*)d_in[11];
  const float* bq = (const float*)d_in[12];
  const float* Wk = (const float*)d_in[13];
  const float* bk = (const float*)d_in[14];
  const float* Wv = (const float*)d_in[15];
  const float* bv = (const float*)d_in[16];
  const float* Wp = (const float*)d_in[17];
  const float* bp = (const float*)d_in[18];
  const float* rel_pe = (const float*)d_in[19];

  float* ws = (float*)d_ws;
  const size_t S = (size_t)B_ * T_ * C_;   // 16.78M floats per tensor
  float* qws   = ws;
  float* kws   = ws + S;
  float* vws   = ws + 2 * S;
  float* attws = ws + 3 * S;

  dim3 gA(T_ / 128, C_ / 64, B_ * 3);
  qkv_wmma_kernel<<<gA, 256, 0, stream>>>(x, dw_q, dw_k, dw_v,
                                          bnq_s, bnq_b, bnk_s, bnk_b, bnv_s, bnv_b,
                                          Wq, bq, Wk, bk, Wv, bv,
                                          qws, kws, vws);

  dim3 gB(T_ / QB_, H_, B_);
  local_attn_kernel<<<gB, QB_, 0, stream>>>(qws, kws, vws, mask, rel_pe, attws);

  dim3 gC(T_ / 128, C_ / 64, B_);
  proj_wmma_kernel<<<gC, 256, 0, stream>>>(attws, Wp, bp, mask, (float*)d_out);
}
